// LSTMDecoder_82781199663559
// MI455X (gfx1250) — matrix operand order repaired and checked by reading, not compiled, not run on hardware
//
#include <hip/hip_runtime.h>
#include <hip/hip_bf16.h>
#include <math.h>

// Problem constants (match reference).
#define VOCAB_N 32000
#define EMBED_N 512
#define HID_N   512
#define BATCH_N 64
#define GATES_N 2048   // 4*HID
#define KCAT_N  1024   // EMBED + HID

typedef __attribute__((ext_vector_type(16))) __bf16 bf16x16;
typedef __attribute__((ext_vector_type(8)))  __bf16 bf16x8;
typedef __attribute__((ext_vector_type(8)))  float  f32x8;

// ---------------- WMMA fragment loaders (CDNA5 16-bit layouts) ----------------
// A matrix 16x32 (bf16): lane half h, elements e=0..7 -> K = h*8 + e,
// e=8..15 -> K = 16 + h*8 + (e-8). Two contiguous 16B runs.
__device__ inline bf16x16 load_a_frag(const __bf16* base, int half) {
  union { bf16x16 f; bf16x8 h[2]; } u;
  u.h[0] = *(const bf16x8*)(base + half * 8);
  u.h[1] = *(const bf16x8*)(base + 16 + half * 8);
  return u.f;
}
// B matrix 32x16 (bf16): lane half h, elements e=0..7 -> K = h*8 + e,
// e=8..15 -> K = 16 + h*8 + (e-8). Two contiguous 16B runs.
__device__ inline bf16x16 load_b_frag(const __bf16* base, int half) {
  union { bf16x16 f; bf16x8 h[2]; } u;
  u.h[0] = *(const bf16x8*)(base + half * 8);
  u.h[1] = *(const bf16x8*)(base + 16 + half * 8);
  return u.f;
}

__device__ inline float sigmoidf_(float x) { return 1.0f / (1.0f + __expf(-x)); }

// ---------------- one-time prep kernels ----------------
__global__ __launch_bounds__(256) void cvt_bf16_kernel(const float* __restrict__ src,
                                                       __bf16* __restrict__ dst, int n) {
  int i = blockIdx.x * 256 + threadIdx.x;
  if (i < n) dst[i] = (__bf16)src[i];
}

__global__ __launch_bounds__(256) void build_wcat_kernel(const float* __restrict__ wih,
                                                         const float* __restrict__ whh,
                                                         __bf16* __restrict__ wcat) {
  int idx = blockIdx.x * 256 + threadIdx.x;           // 0 .. 2048*1024-1
  if (idx >= GATES_N * KCAT_N) return;
  int n = idx >> 10, k = idx & (KCAT_N - 1);
  float v = (k < EMBED_N) ? wih[n * EMBED_N + k] : whh[n * HID_N + (k - EMBED_N)];
  wcat[idx] = (__bf16)v;
}

__global__ __launch_bounds__(256) void bias_kernel(const float* __restrict__ bih,
                                                   const float* __restrict__ bhh,
                                                   float* __restrict__ bias) {
  int i = blockIdx.x * 256 + threadIdx.x;
  if (i < GATES_N) bias[i] = bih[i] + bhh[i];
}

__global__ __launch_bounds__(256) void init_state_kernel(const float* __restrict__ hidden,
                                                         const float* __restrict__ cell,
                                                         const int* __restrict__ start_tok,
                                                         float* __restrict__ c,
                                                         __bf16* __restrict__ xh,
                                                         int* __restrict__ tok) {
  int idx = blockIdx.x * 256 + threadIdx.x;           // 0 .. 64*512-1
  if (idx >= BATCH_N * HID_N) return;
  int b = idx >> 9, j = idx & (HID_N - 1);
  c[idx] = cell[idx];
  xh[b * KCAT_N + EMBED_N + j] = (__bf16)hidden[idx];
  if (idx < BATCH_N) tok[idx] = start_tok[0];
}

// ---------------- per-step kernels ----------------
__global__ __launch_bounds__(256) void gather_x_kernel(const float* __restrict__ emb,
                                                       const int* __restrict__ tok,
                                                       __bf16* __restrict__ xh) {
  int idx = blockIdx.x * 256 + threadIdx.x;           // 0 .. 64*512-1
  if (idx >= BATCH_N * EMBED_N) return;
  int b = idx >> 9, j = idx & (EMBED_N - 1);
  xh[b * KCAT_N + j] = (__bf16)emb[(size_t)tok[b] * EMBED_N + j];
}

// gates[64,2048] = xh[64,1024] @ wcat^T[1024,2048] + bias   (bf16 WMMA, f32 acc)
// 128 N-tiles; one wave per N-tile, 4 M-tiles each.
// K loop fully unrolled (32 chunks): SSA registers per chunk -> no copies/NOPs,
// scheduler hoists clause'd loads ahead of the WMMAs.
__global__ __launch_bounds__(256) void gates_wmma_kernel(const __bf16* __restrict__ xh,
                                                         const __bf16* __restrict__ wcat,
                                                         const float* __restrict__ bias,
                                                         float* __restrict__ gates) {
  int wave = (blockIdx.x * 256 + threadIdx.x) >> 5;   // 0..127 == ntile
  int lane = threadIdx.x & 31;
  int half = lane >> 4, lx = lane & 15;
  f32x8 acc[4] = {f32x8{0}, f32x8{0}, f32x8{0}, f32x8{0}};
  const __bf16* brow = wcat + (size_t)(wave * 16 + lx) * KCAT_N;
  const __bf16* arow = xh + (size_t)lx * KCAT_N;      // + mt*16*KCAT_N as const offset

#pragma unroll
  for (int kk = 0; kk < KCAT_N; kk += 32) {
    bf16x16 b = load_b_frag(brow + kk, half);
    bf16x16 a[4];
#pragma unroll
    for (int mt = 0; mt < 4; ++mt)
      a[mt] = load_a_frag(arow + mt * 16 * KCAT_N + kk, half);
#pragma unroll
    for (int mt = 0; mt < 4; ++mt)
      acc[mt] = __builtin_amdgcn_wmma_f32_16x16x32_bf16(
          false, a[mt], false, b, (short)0, acc[mt], false, false);
  }

  int col = wave * 16 + lx;
  float bv = bias[col];
#pragma unroll
  for (int mt = 0; mt < 4; ++mt)
#pragma unroll
    for (int r = 0; r < 8; ++r) {
      int row = mt * 16 + r + half * 8;
      gates[row * GATES_N + col] = acc[mt][r] + bv;
    }
}

__global__ __launch_bounds__(256) void lstm_update_kernel(const float* __restrict__ gates,
                                                          float* __restrict__ c,
                                                          __bf16* __restrict__ xh) {
  int idx = blockIdx.x * 256 + threadIdx.x;           // 0 .. 64*512-1
  if (idx >= BATCH_N * HID_N) return;
  int b = idx >> 9, j = idx & (HID_N - 1);
  const float* g = gates + b * GATES_N;
  float ig = sigmoidf_(g[j]);
  float fg = sigmoidf_(g[HID_N + j]);
  float gg = tanhf(g[2 * HID_N + j]);
  float og = sigmoidf_(g[3 * HID_N + j]);
  float cn = fg * c[idx] + ig * gg;
  c[idx] = cn;
  xh[b * KCAT_N + EMBED_N + j] = (__bf16)(og * tanhf(cn));
}

// logits[64,32000] = h[64,512] @ fcw^T[512,32000] + fc_b, written to d_out[b,t,:].
// Per block: stage h (64KB) into LDS with CDNA5 async-to-LDS copies; each wave
// runs one N-tile x 4 M-tiles; K loop fully unrolled (16 chunks); B stream from
// L2, A fragments from LDS (ds_load_b128).
__global__ __launch_bounds__(256) void logits_wmma_kernel(const __bf16* __restrict__ xh,
                                                          const __bf16* __restrict__ fcw,
                                                          const float* __restrict__ fcb,
                                                          float* __restrict__ out,
                                                          int t, int T) {
  __shared__ __align__(16) __bf16 sh_h[BATCH_N * HID_N];   // 64 KB

  // ---- async stage: h = xh[:,512:1024] -> LDS (4096 x 16B chunks) ----
  {
    int tid = threadIdx.x;
#pragma unroll
    for (int i = 0; i < 16; ++i) {
      int chunk = tid * 16 + i;                 // 0..4095
      int row = chunk >> 6;                     // 64 chunks of 8 elems per row
      int col8 = chunk & 63;
      const __bf16* g = xh + (size_t)row * KCAT_N + EMBED_N + col8 * 8;
      // ISA 10.2: flat LDS address truncated to 32 bits == raw LDS byte offset.
      unsigned lds_addr = (unsigned)(size_t)&sh_h[row * HID_N + col8 * 8];
      unsigned long long gaddr = (unsigned long long)(size_t)g;
      asm volatile("global_load_async_to_lds_b128 %0, %1, off"
                   :: "v"(lds_addr), "v"(gaddr) : "memory");
    }
    asm volatile("s_wait_asynccnt 0x0" ::: "memory");
    __syncthreads();
  }

  int wave = (blockIdx.x * 256 + threadIdx.x) >> 5;   // 0..1999 == ntile
  int lane = threadIdx.x & 31;
  int half = lane >> 4, lx = lane & 15;
  f32x8 acc[4] = {f32x8{0}, f32x8{0}, f32x8{0}, f32x8{0}};
  const __bf16* brow = fcw + (size_t)(wave * 16 + lx) * HID_N;
  const __bf16* alds = &sh_h[lx * HID_N];             // + mt*16*HID_N as const offset

  __builtin_prefetch(brow + 256, 0, 0);               // global_prefetch_b8

#pragma unroll
  for (int kk = 0; kk < HID_N; kk += 32) {
    bf16x16 b = load_b_frag(brow + kk, half);
#pragma unroll
    for (int mt = 0; mt < 4; ++mt)
      acc[mt] = __builtin_amdgcn_wmma_f32_16x16x32_bf16(
          false, load_a_frag(alds + mt * 16 * HID_N + kk, half),
          false, b, (short)0, acc[mt], false, false);
  }

  int col = wave * 16 + lx;
  float bv = fcb[col];
#pragma unroll
  for (int mt = 0; mt < 4; ++mt)
#pragma unroll
    for (int r = 0; r < 8; ++r) {
      int row = mt * 16 + r + half * 8;
      out[(size_t)row * T * VOCAB_N + (size_t)t * VOCAB_N + col] = acc[mt][r] + bv;
    }
}

__global__ __launch_bounds__(256) void argmax_kernel(const float* __restrict__ out,
                                                     int t, int T, int* __restrict__ tok) {
  int b = blockIdx.x;
  const float* row = out + (size_t)b * T * VOCAB_N + (size_t)t * VOCAB_N;
  float best = -INFINITY;
  int bi = 0x7fffffff;
  for (int v = threadIdx.x; v < VOCAB_N; v += 256) {
    float x = row[v];
    if (x > best) { best = x; bi = v; }
  }
  __shared__ float sv[256];
  __shared__ int si[256];
  sv[threadIdx.x] = best;
  si[threadIdx.x] = bi;
  __syncthreads();
  for (int s = 128; s > 0; s >>= 1) {
    if (threadIdx.x < s) {
      float ov = sv[threadIdx.x + s];
      int oi = si[threadIdx.x + s];
      if (ov > sv[threadIdx.x] || (ov == sv[threadIdx.x] && oi < si[threadIdx.x])) {
        sv[threadIdx.x] = ov;
        si[threadIdx.x] = oi;
      }
    }
    __syncthreads();
  }
  if (threadIdx.x == 0) tok[b] = si[0];
}

// ---------------- launcher ----------------
extern "C" void kernel_launch(void* const* d_in, const int* in_sizes, int n_in,
                              void* d_out, int out_size, void* d_ws, size_t ws_size,
                              hipStream_t stream) {
  const float* hidden = (const float*)d_in[0];   // [1,B,H]
  const float* cell   = (const float*)d_in[1];   // [1,B,H]
  /* d_in[2] = max_length (device scalar; T recovered from out_size) */
  const int*   start  = (const int*)d_in[3];
  const float* emb    = (const float*)d_in[4];   // [V,E]
  const float* wih    = (const float*)d_in[5];   // [4H,E]
  const float* whh    = (const float*)d_in[6];   // [4H,H]
  const float* bih    = (const float*)d_in[7];
  const float* bhh    = (const float*)d_in[8];
  const float* fcw    = (const float*)d_in[9];   // [V,H]
  const float* fcb    = (const float*)d_in[10];
  float* out = (float*)d_out;

  int T = out_size / (BATCH_N * VOCAB_N);        // 64

  // Carve workspace (256B-aligned slices). ~38 MB total.
  char* p = (char*)d_ws;
  auto carve = [&](size_t bytes) {
    void* r = (void*)p;
    p += (bytes + 255) & ~(size_t)255;
    return r;
  };
  __bf16* fcw_bf = (__bf16*)carve((size_t)VOCAB_N * HID_N * 2);
  __bf16* wcat   = (__bf16*)carve((size_t)GATES_N * KCAT_N * 2);
  float*  bias   = (float*)carve(GATES_N * 4);
  __bf16* xh     = (__bf16*)carve(BATCH_N * KCAT_N * 2);
  float*  c      = (float*)carve(BATCH_N * HID_N * 4);
  float*  gates  = (float*)carve(BATCH_N * GATES_N * 4);
  int*    tok    = (int*)carve(BATCH_N * 4);

  // One-time prep (per launch; amortized over T steps).
  {
    int n = VOCAB_N * HID_N;
    cvt_bf16_kernel<<<(n + 255) / 256, 256, 0, stream>>>(fcw, fcw_bf, n);
    build_wcat_kernel<<<(GATES_N * KCAT_N + 255) / 256, 256, 0, stream>>>(wih, whh, wcat);
    bias_kernel<<<(GATES_N + 255) / 256, 256, 0, stream>>>(bih, bhh, bias);
    init_state_kernel<<<(BATCH_N * HID_N + 255) / 256, 256, 0, stream>>>(
        hidden, cell, start, c, xh, tok);
  }

  for (int t = 0; t < T; ++t) {
    gather_x_kernel<<<(BATCH_N * EMBED_N + 255) / 256, 256, 0, stream>>>(emb, tok, xh);
    gates_wmma_kernel<<<(GATES_N / 16) / 8, 256, 0, stream>>>(xh, wcat, bias, gates);
    lstm_update_kernel<<<(BATCH_N * HID_N + 255) / 256, 256, 0, stream>>>(gates, c, xh);
    logits_wmma_kernel<<<(VOCAB_N / 16) / 8, 256, 0, stream>>>(xh, fcw_bf, fcb, out, t, T);
    argmax_kernel<<<BATCH_N, 256, 0, stream>>>(out, t, T, tok);
  }
}
